// SelfAttention_1365799600775
// MI455X (gfx1250) — matrix operand order repaired and checked by reading, not compiled, not run on hardware
//
#include <hip/hip_runtime.h>
#include <hip/hip_bf16.h>

typedef __attribute__((ext_vector_type(16))) _Float16 v16h;
typedef __attribute__((ext_vector_type(8)))  _Float16 v8h;
typedef __attribute__((ext_vector_type(8)))  float    v8f;

#define B_  4
#define S_  2048
#define D_  1024
#define H_  16
#define DH_ 64
#define TD_ 3072   // 3*D

// ---------- WMMA fragment loaders (per CDNA5 ISA layouts) ----------
// A 16x32 f16: lane(0-15)=row M; lanes<16 hold K {0..7,16..23}, lanes>=16 hold {8..15,24..31}
__device__ __forceinline__ v16h load_a_frag(const _Float16* p) {
    v8h lo = *(const v8h*)(p);        // K koff .. koff+7
    v8h hi = *(const v8h*)(p + 16);   // K koff+16 .. koff+23
    v16h f;
#pragma unroll
    for (int i = 0; i < 8; ++i) { f[i] = lo[i]; f[i + 8] = hi[i]; }
    return f;
}
// B 32x16 f16: lane(0-15)=col N; lanes<16 hold K {0..7,16..23}, lanes>=16 hold {8..15,24..31}
__device__ __forceinline__ v16h load_b_frag(const _Float16* p) {
    v8h lo = *(const v8h*)(p);        // K koff .. koff+7
    v8h hi = *(const v8h*)(p + 16);   // K koff+16 .. koff+23
    v16h f;
#pragma unroll
    for (int i = 0; i < 8; ++i) { f[i] = lo[i]; f[i + 8] = hi[i]; }
    return f;
}
#define WMMA_F16(a, b, c) \
    __builtin_amdgcn_wmma_f32_16x16x32_f16(false, (a), false, (b), (short)0, (c), false, false)

// ---------- conversion / repack kernels ----------
__global__ void cvt_f32_to_f16(const float* __restrict__ in, _Float16* __restrict__ out, int n) {
    int i = blockIdx.x * blockDim.x + threadIdx.x;
    if (i < n) out[i] = (_Float16)in[i];
}
// wT[n*K + k] = w[k*N + n]  (f32 -> f16, transpose so B-fragments are contiguous)
__global__ void cvt_transpose(const float* __restrict__ w, _Float16* __restrict__ wT, int N, int K) {
    int i = blockIdx.x * blockDim.x + threadIdx.x;
    if (i >= N * K) return;
    int k = i & (K - 1);
    int n = i >> 10;                  // K == 1024 always here
    wT[(size_t)n * K + k] = (_Float16)w[(size_t)k * N + n];
}
// vT[((b*H+h)*DH + dh)*S + s] = qkv[(b*S+s)*3D + 2D + h*DH + dh]
__global__ void repack_v(const _Float16* __restrict__ qkv, _Float16* __restrict__ vT) {
    int i = blockIdx.x * blockDim.x + threadIdx.x;   // over B*H*DH*S
    int s = i & (S_ - 1);
    int t = i >> 11;
    int dh = t & (DH_ - 1);
    t >>= 6;
    int h = t & (H_ - 1);
    int b = t >> 4;
    vT[i] = qkv[(size_t)(b * S_ + s) * TD_ + 2 * D_ + h * DH_ + dh];
}

// ---------- WMMA GEMM: C[M,N] = A[M,K] * BT[N,K]^T + bias ----------
// one wave computes a 32x64 tile: 8 WMMAs per k-step on (2 A + 4 B) fragments,
// register double-buffered; unroll 2 => ping-pong buffers, no rotation moves.
template <bool F32OUT>
__global__ __launch_bounds__(256, 1) void
wmma_gemm(const _Float16* __restrict__ A, const _Float16* __restrict__ BT,
          const float* __restrict__ bias, void* __restrict__ Cout,
          int M, int N, int K) {
    const int lane = threadIdx.x & 31;
    const int wave = blockIdx.x * (blockDim.x >> 5) + (threadIdx.x >> 5);
    const int tilesN = N >> 6;
    const int wm = wave / tilesN, wn = wave % tilesN;
    const int m0 = wm << 5, n0 = wn << 6;
    if (m0 >= M) return;

    const int r     = lane & 15;
    const int half  = lane >> 4;
    const int koffA = half * 8;
    const int koffB = half * 8;

    const _Float16* ap0 = A  + (size_t)(m0 + r) * K + koffA;
    const _Float16* ap1 = A  + (size_t)(m0 + 16 + r) * K + koffA;
    const _Float16* bp  = BT + (size_t)(n0 + r) * K + koffB;

    v8f acc[2][4] = {};

    // prologue: load k-step 0 fragments
    v16h a0 = load_a_frag(ap0);
    v16h a1 = load_a_frag(ap1);
    v16h bf[4];
#pragma unroll
    for (int j = 0; j < 4; ++j) bf[j] = load_b_frag(bp + (size_t)(j * 16) * K);

#pragma unroll 2
    for (int kb = 32; kb < K; kb += 32) {
        // issue next k-step's loads first (latency hidden behind the 8 WMMAs below)
        __builtin_prefetch(ap0 + kb + 128, 0, 1);
        __builtin_prefetch(bp + kb + 128, 0, 1);
        v16h a0n = load_a_frag(ap0 + kb);
        v16h a1n = load_a_frag(ap1 + kb);
        v16h bfn[4];
#pragma unroll
        for (int j = 0; j < 4; ++j) bfn[j] = load_b_frag(bp + (size_t)(j * 16) * K + kb);

#pragma unroll
        for (int j = 0; j < 4; ++j) {
            acc[0][j] = WMMA_F16(a0, bf[j], acc[0][j]);
            acc[1][j] = WMMA_F16(a1, bf[j], acc[1][j]);
        }
        a0 = a0n; a1 = a1n;
#pragma unroll
        for (int j = 0; j < 4; ++j) bf[j] = bfn[j];
    }
    // epilogue k-step
#pragma unroll
    for (int j = 0; j < 4; ++j) {
        acc[0][j] = WMMA_F16(a0, bf[j], acc[0][j]);
        acc[1][j] = WMMA_F16(a1, bf[j], acc[1][j]);
    }

    const int col = lane & 15;
#pragma unroll
    for (int j = 0; j < 4; ++j) {
        const int c = n0 + j * 16 + col;
        const float bv = bias[c];
#pragma unroll
        for (int mi = 0; mi < 2; ++mi) {
#pragma unroll
            for (int i = 0; i < 8; ++i) {
                const int row = m0 + mi * 16 + i + 8 * half;  // C layout: vgpr i -> M=i (+8 hi half)
                if (F32OUT) ((float*)Cout)[(size_t)row * N + c] = acc[mi][j][i] + bv;
                else        ((_Float16*)Cout)[(size_t)row * N + c] = (_Float16)(acc[mi][j][i] + bv);
            }
        }
    }
}

// ---------- flash attention: one wave per (b, h, 16-query tile) ----------
// K fragments double-buffered across key tiles; V fragments hoisted over softmax;
// unroll 2 => ping-pong K buffers without rotation moves.
__global__ __launch_bounds__(256, 1) void
flash_attn(const _Float16* __restrict__ qkv, const _Float16* __restrict__ vT,
           _Float16* __restrict__ ctx, const int* __restrict__ causal_flag) {
    __shared__ _Float16 lds_p[8][16 * 32];               // per-wave 16x32 probability tile

    const int causal = *causal_flag;
    const int lane  = threadIdx.x & 31;
    const int wslot = threadIdx.x >> 5;
    const int wave  = blockIdx.x * 8 + wslot;            // B*H*(S/16) = 8192 waves
    const int qt = wave & 127;
    const int h  = (wave >> 7) & (H_ - 1);
    const int b  = wave >> 11;
    const int q0 = qt << 4;

    const int r     = lane & 15;
    const int half  = lane >> 4;
    const int koffA = half * 8;
    const int koffB = half * 8;

    // Q fragments (A operand), rows q0..q0+15, K-dim = dh (64 -> two frags)
    const _Float16* qb = qkv + (size_t)(b * S_ + q0 + r) * TD_ + h * DH_ + koffA;
    const v16h qf0 = load_a_frag(qb);
    const v16h qf1 = load_a_frag(qb + 32);

    // K base (B operand of Q*K^T): column n = key index, contiguous along dh
    const _Float16* kbase = qkv + (size_t)(b * S_ + r) * TD_ + D_ + h * DH_ + koffB;
    // V base (B operand of P*V): vT rows are dh, contiguous along key
    const _Float16* vbase = vT + (size_t)((b * H_ + h) * DH_ + r) * S_ + koffB;

    v8f acc[4] = {};
    float m[8], l[8];
#pragma unroll
    for (int i = 0; i < 8; ++i) { m[i] = -3.0e38f; l[i] = 0.0f; }

    int kend = causal ? (q0 + 16) : S_;
    kend = (kend + 31) & ~31;
    if (kend > S_) kend = S_;

    // prologue: K fragments for tile 0
    v16h k0a = load_b_frag(kbase);
    v16h k0b = load_b_frag(kbase + 32);
    v16h k1a = load_b_frag(kbase + (size_t)16 * TD_);
    v16h k1b = load_b_frag(kbase + (size_t)16 * TD_ + 32);

#pragma unroll 2
    for (int kb = 0; kb < kend; kb += 32) {
        // ---- scores from resident K fragments (loaded a full iteration ago) ----
        v8f s0 = {}, s1 = {};
        s0 = WMMA_F16(qf0, k0a, s0);
        s0 = WMMA_F16(qf1, k0b, s0);
        s1 = WMMA_F16(qf0, k1a, s1);
        s1 = WMMA_F16(qf1, k1b, s1);

        // ---- issue next K tile + current V tile loads NOW; they retire during
        //      softmax (~50 VALU + shuffles). Branch-free: clamp on last iter. ----
        const int kbn = (kb + 32 < kend) ? (kb + 32) : kb;
        const _Float16* kpn = kbase + (size_t)kbn * TD_;
        __builtin_prefetch(kpn + (size_t)32 * TD_, 0, 1);
        v16h nk0a = load_b_frag(kpn);
        v16h nk0b = load_b_frag(kpn + 32);
        v16h nk1a = load_b_frag(kpn + (size_t)16 * TD_);
        v16h nk1b = load_b_frag(kpn + (size_t)16 * TD_ + 32);

        const _Float16* vp = vbase + kb;
        v16h bv0 = load_b_frag(vp);
        v16h bv1 = load_b_frag(vp + (size_t)16 * S_);
        v16h bv2 = load_b_frag(vp + (size_t)32 * S_);
        v16h bv3 = load_b_frag(vp + (size_t)48 * S_);

        const float sc = 0.125f;                          // 1/sqrt(DH)
#pragma unroll
        for (int i = 0; i < 8; ++i) { s0[i] *= sc; s1[i] *= sc; }
        if (causal && (kb + 31 > q0)) {
#pragma unroll
            for (int i = 0; i < 8; ++i) {
                const int row = q0 + i + 8 * half;
                if (kb + r      > row) s0[i] = -3.0e38f;
                if (kb + 16 + r > row) s1[i] = -3.0e38f;
            }
        }
        // ---- online softmax (rows 0-7 live in lanes 0-15, rows 8-15 in lanes 16-31) ----
#pragma unroll
        for (int i = 0; i < 8; ++i) {
            float t = fmaxf(s0[i], s1[i]);
            t = fmaxf(t, __shfl_xor(t, 1, 16));
            t = fmaxf(t, __shfl_xor(t, 2, 16));
            t = fmaxf(t, __shfl_xor(t, 4, 16));
            t = fmaxf(t, __shfl_xor(t, 8, 16));
            const float mn = fmaxf(m[i], t);
            const float sf = __expf(m[i] - mn);
            const float p0 = __expf(s0[i] - mn);
            const float p1 = __expf(s1[i] - mn);
            float rs = p0 + p1;
            rs += __shfl_xor(rs, 1, 16);
            rs += __shfl_xor(rs, 2, 16);
            rs += __shfl_xor(rs, 4, 16);
            rs += __shfl_xor(rs, 8, 16);
            l[i] = l[i] * sf + rs;
            m[i] = mn;
            acc[0][i] *= sf; acc[1][i] *= sf; acc[2][i] *= sf; acc[3][i] *= sf;
            const int row = i + 8 * half;                 // C-layout -> row-major LDS
            lds_p[wslot][row * 32 + r]      = (_Float16)p0;
            lds_p[wslot][row * 32 + 16 + r] = (_Float16)p1;
        }
        asm volatile("s_wait_dscnt 0" ::: "memory");      // cross-lane LDS RAW within wave
        // ---- P as A fragment; V fragments already resident ----
        const v16h pa = load_a_frag(&lds_p[wslot][r * 32 + koffA]);
        acc[0] = WMMA_F16(pa, bv0, acc[0]);
        acc[1] = WMMA_F16(pa, bv1, acc[1]);
        acc[2] = WMMA_F16(pa, bv2, acc[2]);
        acc[3] = WMMA_F16(pa, bv3, acc[3]);
        // rotate K double-buffer (copy-propagated away under unroll 2)
        k0a = nk0a; k0b = nk0b; k1a = nk1a; k1b = nk1b;
    }
    // ---- normalize and store context (f16, [B*S, D] row-major) ----
#pragma unroll
    for (int i = 0; i < 8; ++i) {
        const float inv = 1.0f / l[i];
        const int row = q0 + i + 8 * half;
        const size_t base = (size_t)(b * S_ + row) * D_ + h * DH_;
#pragma unroll
        for (int j = 0; j < 4; ++j)
            ctx[base + j * 16 + r] = (_Float16)(acc[j][i] * inv);
    }
}

// ---------- launcher ----------
extern "C" void kernel_launch(void* const* d_in, const int* in_sizes, int n_in,
                              void* d_out, int out_size, void* d_ws, size_t ws_size,
                              hipStream_t stream) {
    const float* x     = (const float*)d_in[0];
    const float* w_in  = (const float*)d_in[1];
    const float* b_in  = (const float*)d_in[2];
    const float* w_out = (const float*)d_in[3];
    const float* b_out = (const float*)d_in[4];
    const int*   cmask = (const int*)d_in[5];

    const size_t nX   = (size_t)B_ * S_ * D_;        // 8,388,608
    const size_t nWin = (size_t)D_ * TD_;            // 3,145,728
    const size_t nWo  = (size_t)D_ * D_;             // 1,048,576
    const size_t nQKV = (size_t)B_ * S_ * TD_;       // 25,165,824
    const size_t nV   = (size_t)B_ * H_ * DH_ * S_;  // 8,388,608

    _Float16* x_h    = (_Float16*)d_ws;
    _Float16* w_inT  = x_h   + nX;
    _Float16* w_outT = w_inT + nWin;
    _Float16* qkv    = w_outT + nWo;
    _Float16* vT     = qkv   + nQKV;
    _Float16* ctx    = vT    + nV;                   // total ~104 MB

    const int T = 256;
    cvt_f32_to_f16<<<(int)(nX + T - 1) / T, T, 0, stream>>>(x, x_h, (int)nX);
    cvt_transpose<<<(int)(nWin + T - 1) / T, T, 0, stream>>>(w_in, w_inT, TD_, D_);
    cvt_transpose<<<(int)(nWo + T - 1) / T, T, 0, stream>>>(w_out, w_outT, D_, D_);

    // QKV projection: M=8192, N=3072, K=1024 -> 32x64 tiles -> 12288 waves -> 1536 blocks
    {
        int waves = (B_ * S_ / 32) * (TD_ / 64);
        wmma_gemm<false><<<waves / 8, T, 0, stream>>>(x_h, w_inT, b_in, (void*)qkv,
                                                      B_ * S_, TD_, D_);
    }
    repack_v<<<(int)(nV + T - 1) / T, T, 0, stream>>>(qkv, vT);

    // attention: 8192 waves -> 1024 blocks
    flash_attn<<<(B_ * H_ * (S_ / 16)) / 8, T, 0, stream>>>(qkv, vT, ctx, cmask);

    // output projection: M=8192, N=1024, K=1024 -> 32x64 tiles -> 4096 waves -> 512 blocks
    {
        int waves = (B_ * S_ / 32) * (D_ / 64);
        wmma_gemm<true><<<waves / 8, T, 0, stream>>>(ctx, w_outT, b_out, d_out,
                                                     B_ * S_, D_, D_);
    }
}
